// RQCodebook_62586263437357
// MI455X (gfx1250) — matrix operand order repaired and checked by reading, not compiled, not run on hardware
//
#include <hip/hip_runtime.h>
#include <hip/hip_bf16.h>
#include <stdint.h>

// ---------------------------------------------------------------------------
// Residual VQ, fused over all 4 levels. Compute-bound (275 GFLOP vs 140 MB),
// so scores use split-f16 WMMA (h+l decomposition, 3 passes of
// v_wmma_f32_16x16x32_f16, f32 accumulate) ~ fp32-accurate, ~2.7x faster than
// the native f32 WMMA path. Three independent accumulators break the
// WMMA->WMMA RAW chain; tile loop kept at unroll 1 so live VGPRs stay < 256
// (no s_set_vgpr_msb churn, 4+ waves/SIMD). Gathers use exact f32 codebook.
//   residual  [65536, 256] f32   (d_in[0])
//   codebooks [4, 2048, 256] f32 (d_in[1])
//   out: quantized [65536,256] f32  ++  codes [65536,4] i32 (bit-stored)
// ---------------------------------------------------------------------------

#define RQ_B 65536
#define RQ_D 256
#define RQ_K 2048
#define RQ_L 4
#define RQ_NT (RQ_K / 16)   // 128 code tiles of 16 per level
#define RQ_NC (RQ_D / 32)   // 8 wmma k-chunks of 32 over D

typedef _Float16 v16h __attribute__((ext_vector_type(16)));
typedef _Float16 v8h __attribute__((ext_vector_type(8)));
typedef float v4f __attribute__((ext_vector_type(4)));
typedef float v8f __attribute__((ext_vector_type(8)));

// Async-LDS builtin pointer types (AS1 global -> AS3 LDS, 16-byte int vectors)
typedef int rq_v4i __attribute__((vector_size(16)));
typedef __attribute__((address_space(1))) rq_v4i* rq_gptr;
typedef __attribute__((address_space(3))) rq_v4i* rq_lptr;

#if defined(__has_builtin)
#if __has_builtin(__builtin_amdgcn_global_load_async_to_lds_b128)
#define USE_ASYNC_LDS 1
#endif
#endif
#ifndef USE_ASYNC_LDS
#define USE_ASYNC_LDS 0
#endif

__device__ __forceinline__ void wait_async_all() {
#if USE_ASYNC_LDS
#if __has_builtin(__builtin_amdgcn_s_wait_asynccnt)
  __builtin_amdgcn_s_wait_asynccnt(0);
#else
  asm volatile("s_wait_asynccnt 0" ::: "memory");
#endif
#endif
}

// Copy 8KB (one f16 16x256 tile) global -> LDS with 128 threads.
__device__ __forceinline__ void copy_tile_h(_Float16* dst, const _Float16* src,
                                            int tid) {
#pragma unroll
  for (int it = 0; it < 4; ++it) {
    const int off = (it * 128 + tid) * 8;  // 16 bytes per thread per iter
#if USE_ASYNC_LDS
    __builtin_amdgcn_global_load_async_to_lds_b128(
        (rq_gptr)(src + off), (rq_lptr)(dst + off), 0, 0);
#else
    *(rq_v4i*)(dst + off) = *(const rq_v4i*)(src + off);
#endif
  }
}

// ---------------------------------------------------------------------------
// Prologue: per code row, e2 = sum d^2 (f32) and f16 hi/lo split planes.
// One wave per code row; 8192 rows total.
// ---------------------------------------------------------------------------
__global__ __launch_bounds__(256) void rq_prep_kernel(
    const float* __restrict__ cb, float* __restrict__ e2,
    _Float16* __restrict__ cbh, _Float16* __restrict__ cbl) {
  const int lane = threadIdx.x & 31;
  const int row = blockIdx.x * 8 + (threadIdx.x >> 5);
  if (row >= RQ_L * RQ_K) return;
  const float* p = cb + (size_t)row * RQ_D;
  _Float16* ph = cbh + (size_t)row * RQ_D;
  _Float16* pl = cbl + (size_t)row * RQ_D;
  float s = 0.f;
#pragma unroll
  for (int i = 0; i < RQ_D / 32; ++i) {
    const int c = lane + 32 * i;
    const float v = p[c];
    s += v * v;
    const _Float16 h = (_Float16)v;
    ph[c] = h;
    pl[c] = (_Float16)(v - (float)h);
  }
#pragma unroll
  for (int m = 16; m >= 1; m >>= 1) s += __shfl_xor(s, m, 32);
  if (lane == 0) e2[row] = s;
}

// ---------------------------------------------------------------------------
// Main fused kernel. 4 waves / block, each wave owns 16 rows.
// Residual lives in registers as a f16 hi/lo split in exact WMMA-A layout:
//   lane (l15,hi): owns cols 16r + 8*hi + j  (r=0..15, j=0..7)
//   A chunk ch (cols 32ch..32ch+31): v16h = [run 2ch (8 f16), run 2ch+1]
// ---------------------------------------------------------------------------
__global__ __launch_bounds__(128) void rq_main_kernel(
    const float* __restrict__ res0, const float* __restrict__ cb,
    const _Float16* __restrict__ cbh, const _Float16* __restrict__ cbl,
    const float* __restrict__ e2, float* __restrict__ outq,
    int* __restrict__ outc) {
  __shared__ _Float16 ldsH[2][16 * RQ_D];  // 8KB x2 hi tiles
  __shared__ _Float16 ldsL[2][16 * RQ_D];  // 8KB x2 lo tiles

  const int tid = threadIdx.x;
  const int lane = tid & 31;
  const int wave = tid >> 5;
  const int l15 = lane & 15;
  const int hi = lane >> 4;
  const int myRow = blockIdx.x * 64 + wave * 16 + l15;

  // ---- load residual rows, split into f16 hi/lo A fragments --------------
  v16h ah[RQ_NC], al[RQ_NC];
  {
    const float* rp = res0 + (size_t)myRow * RQ_D + 8 * hi;
#pragma unroll
    for (int r = 0; r < 16; ++r) {
      const v4f x0 = *(const v4f*)(rp + 16 * r);
      const v4f x1 = *(const v4f*)(rp + 16 * r + 4);
      const int ch = r >> 1, eb = (r & 1) * 8;
#pragma unroll
      for (int j = 0; j < 8; ++j) {
        const float v = (j < 4) ? x0[j] : x1[j - 4];
        const _Float16 h = (_Float16)v;
        ah[ch][eb + j] = h;
        al[ch][eb + j] = (_Float16)(v - (float)h);
      }
    }
  }

#pragma unroll 1
  for (int l = 0; l < RQ_L; ++l) {
    const float* cbf = cb + (size_t)l * RQ_K * RQ_D;
    const _Float16* cbhl = cbh + (size_t)l * RQ_K * RQ_D;
    const _Float16* cbll = cbl + (size_t)l * RQ_K * RQ_D;
    const float* e2l = e2 + l * RQ_K;

    float best[8];
    int bidx[8];
#pragma unroll
    for (int j = 0; j < 8; ++j) { best[j] = 3.402823466e38f; bidx[j] = 0; }

    copy_tile_h(ldsH[0], cbhl, tid);
    copy_tile_h(ldsL[0], cbll, tid);
    wait_async_all();
    __syncthreads();

#pragma unroll 1
    for (int t = 0; t < RQ_NT; ++t) {
      const int cur = t & 1;
      if (t + 1 < RQ_NT) {
        copy_tile_h(ldsH[cur ^ 1], cbhl + (t + 1) * 16 * RQ_D, tid);
        copy_tile_h(ldsL[cur ^ 1], cbll + (t + 1) * 16 * RQ_D, tid);
      }

      // 16x16 score tile: dot ~= ah*bh + al*bh + ah*bl (f32 accumulate).
      // Three independent accumulators -> WMMA chains at reuse distance 3.
      v8f cHH = {}, cLH = {}, cHL = {};
      const _Float16* BH = &ldsH[cur][l15 * RQ_D + 8 * hi];
      const _Float16* BL = &ldsL[cur][l15 * RQ_D + 8 * hi];
#pragma unroll
      for (int ch = 0; ch < RQ_NC; ++ch) {
        const v16h bh = __builtin_shufflevector(*(const v8h*)(BH + 32 * ch), *(const v8h*)(BH + 32 * ch + 16), 0, 1, 2, 3, 4, 5, 6, 7, 8, 9, 10, 11, 12, 13, 14, 15);
        const v16h bl = __builtin_shufflevector(*(const v8h*)(BL + 32 * ch), *(const v8h*)(BL + 32 * ch + 16), 0, 1, 2, 3, 4, 5, 6, 7, 8, 9, 10, 11, 12, 13, 14, 15);
        cHH = __builtin_amdgcn_wmma_f32_16x16x32_f16(false, ah[ch], false, bh,
                                                     (short)0, cHH, false, false);
        cLH = __builtin_amdgcn_wmma_f32_16x16x32_f16(false, al[ch], false, bh,
                                                     (short)0, cLH, false, false);
        cHL = __builtin_amdgcn_wmma_f32_16x16x32_f16(false, ah[ch], false, bl,
                                                     (short)0, cHL, false, false);
      }

      // score = ||e||^2 - 2*dot  (||res||^2 constant under argmin)
      const float e2v = e2l[t * 16 + l15];
      const int kk = t * 16 + l15;
#pragma unroll
      for (int j = 0; j < 8; ++j) {
        const float dot = (cHH[j] + cLH[j]) + cHL[j];
        const float sc = e2v - 2.0f * dot;
        if (sc < best[j]) { best[j] = sc; bidx[j] = kk; }
      }

      wait_async_all();
      __syncthreads();
    }

    // Cross-lane argmin within each 16-lane half; ties -> lower index.
#pragma unroll
    for (int j = 0; j < 8; ++j) {
      float bs = best[j];
      int bi = bidx[j];
#pragma unroll
      for (int m = 1; m < 16; m <<= 1) {
        const float os = __shfl_xor(bs, m, 16);
        const int oi = __shfl_xor(bi, m, 16);
        if (os < bs || (os == bs && oi < bi)) { bs = os; bi = oi; }
      }
      bidx[j] = bi;
    }

    // Route winning index to the lane pair owning row l15.
    int kst = 0;
#pragma unroll
    for (int j = 0; j < 8; ++j) {
      const int k0 = __shfl(bidx[j], 0, 32);   // row j
      const int k1 = __shfl(bidx[j], 16, 32);  // row j + 8
      if ((l15 & 7) == j) kst = (l15 >> 3) ? k1 : k0;
    }
    if (hi == 0) outc[myRow * RQ_L + l] = kst;

    // Gather exact f32 code row; update residual split in place.
    const float* qp = cbf + (size_t)kst * RQ_D + 8 * hi;
#pragma unroll
    for (int r = 0; r < 16; ++r) {
      const v4f q0 = *(const v4f*)(qp + 16 * r);
      const v4f q1 = *(const v4f*)(qp + 16 * r + 4);
      const int ch = r >> 1, eb = (r & 1) * 8;
#pragma unroll
      for (int j = 0; j < 8; ++j) {
        float rv = (float)ah[ch][eb + j] + (float)al[ch][eb + j];
        rv -= (j < 4) ? q0[j] : q1[j - 4];
        const _Float16 h = (_Float16)rv;
        ah[ch][eb + j] = h;
        al[ch][eb + j] = (_Float16)(rv - (float)h);
      }
    }
    __syncthreads();
  }

  // quantized = res0 - res_final
  {
    const float* rp = res0 + (size_t)myRow * RQ_D + 8 * hi;
    float* op = outq + (size_t)myRow * RQ_D + 8 * hi;
#pragma unroll
    for (int r = 0; r < 16; ++r) {
      const v4f x0 = *(const v4f*)(rp + 16 * r);
      const v4f x1 = *(const v4f*)(rp + 16 * r + 4);
      const int ch = r >> 1, eb = (r & 1) * 8;
      v4f o0, o1;
#pragma unroll
      for (int j = 0; j < 8; ++j) {
        const float rv = (float)ah[ch][eb + j] + (float)al[ch][eb + j];
        const float o = ((j < 4) ? x0[j] : x1[j - 4]) - rv;
        if (j < 4) o0[j] = o; else o1[j - 4] = o;
      }
      *(v4f*)(op + 16 * r) = o0;
      *(v4f*)(op + 16 * r + 4) = o1;
    }
  }
}

// ---------------------------------------------------------------------------
extern "C" void kernel_launch(void* const* d_in, const int* in_sizes, int n_in,
                              void* d_out, int out_size, void* d_ws, size_t ws_size,
                              hipStream_t stream) {
  (void)in_sizes; (void)n_in; (void)out_size; (void)ws_size;
  const float* residual = (const float*)d_in[0];
  const float* codebooks = (const float*)d_in[1];

  // Workspace: e2 (32KB f32) | cb hi plane (4MB f16) | cb lo plane (4MB f16)
  float* e2 = (float*)d_ws;
  _Float16* cbh = (_Float16*)((char*)d_ws + (size_t)RQ_L * RQ_K * sizeof(float));
  _Float16* cbl = cbh + (size_t)RQ_L * RQ_K * RQ_D;

  float* outq = (float*)d_out;
  int* outc = (int*)d_out + (size_t)RQ_B * RQ_D;

  rq_prep_kernel<<<RQ_L * RQ_K / 8, 256, 0, stream>>>(codebooks, e2, cbh, cbl);
  rq_main_kernel<<<RQ_B / 64, 128, 0, stream>>>(residual, codebooks, cbh, cbl,
                                                e2, outq, outc);
}
